// PoincareConv1d_85280870629533
// MI455X (gfx1250) — matrix operand order repaired and checked by reading, not compiled, not run on hardware
//
#include <hip/hip_runtime.h>
#include <cmath>

typedef __attribute__((ext_vector_type(16))) _Float16 v16h;
typedef __attribute__((ext_vector_type(8)))  _Float16 v8h;
typedef __attribute__((ext_vector_type(8)))  float    v8f;

#define C_IN   256
#define C_OUT  512
#define KW     3
#define LSEQ   2048
#define KDIM   768           // C_IN*KW
#define MTILE  32
#define LDK    776           // 768 + 8 halfs padding
#define KSTEPS (KDIM / 32)   // 24

// ---------------- Stage 1: column-normalize weights, precompute per-output params ----
__global__ __launch_bounds__(256)
void zunit_kernel(const float* __restrict__ wv, const float* __restrict__ wg,
                  const float* __restrict__ mlr,
                  _Float16* __restrict__ zt, float* __restrict__ coshd,
                  float* __restrict__ sinhd, float* __restrict__ wg2)
{
    int j = blockIdx.x;            // output unit 0..511
    int tid = threadIdx.x;         // 256 threads
    __shared__ float red[256];
    float s = 0.f;
    for (int k = tid; k < KDIM; k += 256) {
        float v = wv[(size_t)k * C_OUT + j];
        s += v * v;
    }
    red[tid] = s; __syncthreads();
    for (int off = 128; off > 0; off >>= 1) {
        if (tid < off) red[tid] += red[tid + off];
        __syncthreads();
    }
    float norm = fmaxf(sqrtf(red[0]), 1e-15f);
    float inv = 1.0f / norm;
    for (int k = tid; k < KDIM; k += 256)
        zt[(size_t)j * KDIM + k] = (_Float16)(wv[(size_t)k * C_OUT + j] * inv);
    if (tid == 0) {
        float d = 2.0f * mlr[j];
        coshd[j] = coshf(d);
        sinhd[j] = sinhf(d);
        wg2[j]   = 2.0f * wg[j];
    }
}

// ---------------- Stage 2: per-position channel sum-of-squares ----------------------
__global__ __launch_bounds__(256)
void colsq_kernel(const float* __restrict__ x, float* __restrict__ colsq)
{
    // grid = 16 * (2048/64); each block: one batch, 64 positions
    int blk = blockIdx.x;
    int b   = blk >> 5;                 // /32
    int tb  = (blk & 31) << 6;          // *64
    int tid = threadIdx.x;
    int t   = tb + (tid & 63);
    int cq  = tid >> 6;                 // 0..3
    const float* xb = x + (size_t)b * C_IN * LSEQ;
    float s = 0.f;
    for (int c = cq; c < C_IN; c += 4) {
        float v = xb[(size_t)c * LSEQ + t];
        s += v * v;
    }
    __shared__ float red[4][64];
    red[cq][tid & 63] = s; __syncthreads();
    if (cq == 0)
        colsq[b * LSEQ + t] = red[0][tid & 63] + red[1][tid & 63]
                            + red[2][tid & 63] + red[3][tid & 63];
}

// ---------------- Stage 3: WMMA GEMM + hyperbolic epilogue --------------------------
__global__ __launch_bounds__(256)
void poincare_main_kernel(const float* __restrict__ x, const float* __restrict__ bias,
                          const _Float16* __restrict__ zt, const float* __restrict__ colsq,
                          const float* __restrict__ coshd, const float* __restrict__ sinhd,
                          const float* __restrict__ wg2, float* __restrict__ out,
                          float scale, float scale2)
{
    __shared__ _Float16 Atile[MTILE * LDK];          // ~49.7 KB
    __shared__ float sp_cosh[C_OUT], sp_sinh[C_OUT], sp_wg2[C_OUT], sp_bias[C_OUT];
    __shared__ float rowsq[MTILE], rowacc[MTILE];

    const int b   = blockIdx.x >> 6;                 // 16 batches
    const int l0  = (blockIdx.x & 63) << 5;          // 64 tiles of 32 rows
    const int tid = threadIdx.x;

    // preload per-output params
    for (int i = tid; i < C_OUT; i += 256) {
        sp_cosh[i] = coshd[i]; sp_sinh[i] = sinhd[i];
        sp_wg2[i]  = wg2[i];   sp_bias[i] = bias[i];
    }
    // per-row squared norm of the SCALE*patch (window over colsq), zero row accumulators
    if (tid < MTILE) {
        int l = l0 + tid;
        float s = 0.f;
        #pragma unroll
        for (int kk = 0; kk < KW; ++kk) {
            int li = l + kk - 1;                     // PAD = 1
            if (li >= 0 && li < LSEQ) s += colsq[b * LSEQ + li];
        }
        rowsq[tid]  = scale2 * s;
        rowacc[tid] = 0.f;
    }
    // fill A tile: Atile[m][c*3+kk] = f16(scale * x_pad[b, c, l0+m+kk-1])
    const float* xb = x + (size_t)b * C_IN * LSEQ;
    #pragma unroll 4
    for (int i = 0; i < (MTILE * KDIM) / 256; ++i) {
        int e   = tid + i * 256;
        int c   = e / 96;            // 96 = KW*MTILE
        int rem = e % 96;
        int kk  = rem >> 5;
        int m   = rem & 31;
        int li  = l0 + m + kk - 1;
        float v = (li >= 0 && li < LSEQ) ? xb[(size_t)c * LSEQ + li] : 0.f;
        Atile[m * LDK + c * KW + kk] = (_Float16)(scale * v);
    }
    __syncthreads();

    const int wid  = tid >> 5;
    const int lane = tid & 31;
    const int lm   = lane & 15;
    const int g    = lane >> 4;

    // Per-wave fragment base pointers (k-offset folds into the 24-bit immediate).
    // A (16x32 f16): lane = row M, halves 0..7 at K = g*8.., halves 8..15 at K = 16+g*8..
    const _Float16* aBase0 = &Atile[(lm)      * LDK + g * 8];
    const _Float16* aBase1 = &Atile[(16 + lm) * LDK + g * 8];
    // B (32x16 f16): lane = col N = wid*64 + nt*16 + lm, halves 0..7 at K = g*8.., halves 8..15 at K = 16+g*8..
    const _Float16* bBase[4];
    #pragma unroll
    for (int nt = 0; nt < 4; ++nt)
        bBase[nt] = zt + (size_t)(wid * 64 + nt * 16 + lm) * KDIM + g * 8;

    v8f acc[2][4] = {};
    v16h aC[2], bC[4], aN[2], bN[4];

    // prologue loads for kt = 0
    {
        v8h lo0 = *(const v8h*)(aBase0);      v8h hi0 = *(const v8h*)(aBase0 + 16);
        v8h lo1 = *(const v8h*)(aBase1);      v8h hi1 = *(const v8h*)(aBase1 + 16);
        aC[0] = __builtin_shufflevector(lo0, hi0, 0,1,2,3,4,5,6,7,8,9,10,11,12,13,14,15);
        aC[1] = __builtin_shufflevector(lo1, hi1, 0,1,2,3,4,5,6,7,8,9,10,11,12,13,14,15);
        #pragma unroll
        for (int nt = 0; nt < 4; ++nt) bC[nt] = __builtin_shufflevector(*(const v8h*)(bBase[nt]), *(const v8h*)(bBase[nt] + 16), 0,1,2,3,4,5,6,7,8,9,10,11,12,13,14,15);
    }

    // software-pipelined K loop: prefetch kt+1 before consuming kt
    #pragma unroll 2
    for (int kt = 0; kt < KSTEPS; ++kt) {
        const int kn = (kt + 1) * 32;        // next step's K offset (halfs)
        if (kt + 1 < KSTEPS) {
            v8h lo0 = *(const v8h*)(aBase0 + kn);  v8h hi0 = *(const v8h*)(aBase0 + kn + 16);
            v8h lo1 = *(const v8h*)(aBase1 + kn);  v8h hi1 = *(const v8h*)(aBase1 + kn + 16);
            aN[0] = __builtin_shufflevector(lo0, hi0, 0,1,2,3,4,5,6,7,8,9,10,11,12,13,14,15);
            aN[1] = __builtin_shufflevector(lo1, hi1, 0,1,2,3,4,5,6,7,8,9,10,11,12,13,14,15);
            #pragma unroll
            for (int nt = 0; nt < 4; ++nt) bN[nt] = __builtin_shufflevector(*(const v8h*)(bBase[nt] + kn), *(const v8h*)(bBase[nt] + kn + 16), 0,1,2,3,4,5,6,7,8,9,10,11,12,13,14,15);
        }
        #pragma unroll
        for (int nt = 0; nt < 4; ++nt) {
            acc[0][nt] = __builtin_amdgcn_wmma_f32_16x16x32_f16(
                false, aC[0], false, bC[nt], (short)0, acc[0][nt], false, false);
            acc[1][nt] = __builtin_amdgcn_wmma_f32_16x16x32_f16(
                false, aC[1], false, bC[nt], (short)0, acc[1][nt], false, false);
        }
        aC[0] = aN[0]; aC[1] = aN[1];
        #pragma unroll
        for (int nt = 0; nt < 4; ++nt) bC[nt] = bN[nt];
    }

    // ---- epilogue ----
    // per-row scalars (same for all columns): s = tanh(un)/un, cx2 = tanh(un)^2
    float srow[2][8], c1[2][8], invd[2][8];
    #pragma unroll
    for (int mt = 0; mt < 2; ++mt)
        #pragma unroll
        for (int r = 0; r < 8; ++r) {
            int m   = mt * 16 + g * 8 + r;
            float un = fmaxf(sqrtf(fmaxf(rowsq[m], 1e-30f)), 1e-15f);
            float th = tanhf(un);
            srow[mt][r] = th / un;
            float cx2   = th * th;
            c1[mt][r]   = 1.0f + cx2;
            invd[mt][r] = 1.0f / (1.0f - cx2);
        }

    // y = sinh( 2*g*asinh(num/(1-cx2)) + bias );  accumulate per-row sum(y^2)
    #pragma unroll
    for (int mt = 0; mt < 2; ++mt)
        #pragma unroll
        for (int r = 0; r < 8; ++r) {
            float part = 0.f;
            #pragma unroll
            for (int nt = 0; nt < 4; ++nt) {
                int n = wid * 64 + nt * 16 + lm;
                float dot = srow[mt][r] * acc[mt][nt][r];
                float num = 2.0f * dot * sp_cosh[n] - c1[mt][r] * sp_sinh[n];
                float yv  = sp_wg2[n] * asinhf(num * invd[mt][r]) + sp_bias[n];
                yv = sinhf(yv);
                acc[mt][nt][r] = yv;
                part += yv * yv;
            }
            atomicAdd(&rowacc[mt * 16 + g * 8 + r], part);
        }
    __syncthreads();

    // project into ball and store (C/D layout: lane row m = mt*16 + 8*g + r, col = n)
    #pragma unroll
    for (int mt = 0; mt < 2; ++mt) {
        const int m0 = mt * 16 + g * 8;
        float dsc[8];
        #pragma unroll
        for (int r = 0; r < 8; ++r)
            dsc[r] = 1.0f / (1.0f + sqrtf(1.0f + rowacc[m0 + r]));
        #pragma unroll
        for (int nt = 0; nt < 4; ++nt) {
            int n = wid * 64 + nt * 16 + lm;
            float* op = out + ((size_t)(b * C_OUT + n)) * LSEQ + l0 + m0;
            float4 v0, v1;
            v0.x = acc[mt][nt][0] * dsc[0]; v0.y = acc[mt][nt][1] * dsc[1];
            v0.z = acc[mt][nt][2] * dsc[2]; v0.w = acc[mt][nt][3] * dsc[3];
            v1.x = acc[mt][nt][4] * dsc[4]; v1.y = acc[mt][nt][5] * dsc[5];
            v1.z = acc[mt][nt][6] * dsc[6]; v1.w = acc[mt][nt][7] * dsc[7];
            *(float4*)(op)     = v0;
            *(float4*)(op + 4) = v1;
        }
    }
}

extern "C" void kernel_launch(void* const* d_in, const int* in_sizes, int n_in,
                              void* d_out, int out_size, void* d_ws, size_t ws_size,
                              hipStream_t stream)
{
    const float* x   = (const float*)d_in[0];   // (16, 256, 2048)
    const float* wv  = (const float*)d_in[1];   // (768, 512)
    const float* wg  = (const float*)d_in[2];   // (512,)
    const float* mlr = (const float*)d_in[3];   // (512,)
    const float* bs  = (const float*)d_in[4];   // (512,)
    float* out = (float*)d_out;                 // (16, 512, 2048)

    // workspace layout
    char* ws = (char*)d_ws;
    _Float16* zt  = (_Float16*)ws;                          // 512*768*2 = 786432
    float* colsq  = (float*)(ws + 786432);                  // 16*2048*4 = 131072
    float* coshd  = (float*)(ws + 786432 + 131072);         // 2048
    float* sinhd  = (float*)(ws + 786432 + 131072 + 2048);  // 2048
    float* wg2    = (float*)(ws + 786432 + 131072 + 4096);  // 2048

    // SCALE = exp(lbeta(C_IN*K/2, 0.5) - lbeta(C_IN/2, 0.5)), curvature c = 1
    const double lb1 = lgamma(384.0) + lgamma(0.5) - lgamma(384.5);
    const double lb2 = lgamma(128.0) + lgamma(0.5) - lgamma(128.5);
    const float scale  = (float)exp(lb1 - lb2);
    const float scale2 = scale * scale;

    zunit_kernel<<<C_OUT, 256, 0, stream>>>(wv, wg, mlr, zt, coshd, sinhd, wg2);
    colsq_kernel<<<16 * (LSEQ / 64), 256, 0, stream>>>(x, colsq);
    poincare_main_kernel<<<16 * (LSEQ / MTILE), 256, 0, stream>>>(
        x, bs, zt, colsq, coshd, sinhd, wg2, out, scale, scale2);
}
